// MultiHeadAttention_18210661335293
// MI455X (gfx1250) — compile-verified, matrix operand order checked by reading, not run on hardware
//
#include <hip/hip_runtime.h>
#include <hip/hip_bf16.h>

#define BB 2
#define SS 2048
#define DD 1024
#define HH 16
#define EE 64

typedef __attribute__((ext_vector_type(16))) __bf16 v16bf;
typedef __attribute__((ext_vector_type(8)))  float  v8f;
typedef __attribute__((ext_vector_type(4)))  unsigned int v4u;
typedef __attribute__((ext_vector_type(8)))  int v8i;
typedef __attribute__((ext_vector_type(4)))  int v4i;

union TileU { v16bf v; unsigned int u[8]; };

__device__ __forceinline__ v8f wmma_bf16(v16bf a, v16bf b, v8f c) {
    // 8 args: (neg_a, A, neg_b, B, c_mod, C, reuse_a, reuse_b)
    return __builtin_amdgcn_wmma_f32_16x16x32_bf16(false, a, false, b, (short)0, c, false, false);
}

// Load a 16x32 bf16 tile in WMMA A/B per-lane layout from a row-major matrix
// (global or LDS). Lane l of each half holds row (row0+l); element e maps to
// k = k0 + (e&7) + 16*(e>>3) + 8*half.
__device__ __forceinline__ v16bf load_tile(const __hip_bfloat16* p, int ld, int row0, int k0) {
    const int lane = threadIdx.x & 31;
    const int half = lane >> 4;
    const int l = lane & 15;
    const unsigned int* r =
        (const unsigned int*)(p + (size_t)(row0 + l) * (size_t)ld + (size_t)(k0 + 8 * half));
    TileU t;
    t.u[0] = r[0];  t.u[1] = r[1];  t.u[2] = r[2];  t.u[3] = r[3];   // k0+8h .. +7
    t.u[4] = r[8];  t.u[5] = r[9];  t.u[6] = r[10]; t.u[7] = r[11];  // k0+16+8h .. +23
    return t.v;
}

// ---------------- Tensor Data Mover: 2-D tile Global -> LDS ----------------
// Builds a D# per CDNA5 ISA sec.8 and issues TENSOR_LOAD_TO_LDS.
// data_size = 2 bytes (bf16). LDS row padding for bank-conflict-free reads.
__device__ __forceinline__ unsigned lds_offset(const void* p) {
    // generic shared address: low 32 bits are the LDS byte offset (ISA 10.2 aperture rules)
    return (unsigned)(size_t)p;
}

__device__ __forceinline__ void tdm_load_2d_bf16(
    unsigned lds_off, const void* gptr,
    unsigned tile_d0, unsigned tile_d1,          // elements / rows
    unsigned tensor_d0, unsigned tensor_d1,      // tensor extents (elements)
    unsigned long long stride0,                  // row stride (elements)
    unsigned pad_interval_code, unsigned pad_amount_code)
{
    const unsigned long long ga = (unsigned long long)(size_t)gptr;
    v4u g0;
    g0.x = 1u;                                        // count=1, user descriptor
    g0.y = lds_off;                                   // lds_addr (bytes)
    g0.z = (unsigned)(ga & 0xffffffffu);              // global_addr[31:0]
    g0.w = (unsigned)((ga >> 32) & 0x1ffffffu)        // global_addr[56:32]
         | (2u << 30);                                // type = 2 ("image")
    v8i g1;
    g1[0] = (int)((1u << 16)                          // data_size = 2 bytes
         | (1u << 20)                                 // pad_enable
         | (pad_interval_code << 22)
         | (pad_amount_code << 25));
    g1[1] = (int)((tensor_d0 & 0xffffu) << 16);       // tensor_dim0[15:0] (barrier addr = 0)
    g1[2] = (int)(((tensor_d0 >> 16) & 0xffffu) | ((tensor_d1 & 0xffffu) << 16));
    g1[3] = (int)(((tensor_d1 >> 16) & 0xffffu) | ((tile_d0 & 0xffffu) << 16));
    g1[4] = (int)(tile_d1 & 0xffffu);                 // tile_dim1; tile_dim2 = 0 (2-D)
    g1[5] = (int)(unsigned)(stride0 & 0xffffffffull); // tensor_dim0_stride[31:0]
    g1[6] = (int)(unsigned)((stride0 >> 32) & 0xffffull);
    g1[7] = 0;
    const v4i z4 = {0, 0, 0, 0};
    const v8i z8 = {0, 0, 0, 0, 0, 0, 0, 0};
    // 6-arg toolchain variant: (g0, g1, g2, g3, extra group, cpol)
    __builtin_amdgcn_tensor_load_to_lds(g0, g1, z4, z4, z8, 0);
}

// ---------------- conversion / layout kernels ----------------

__global__ void f32_to_bf16_kernel(const float* __restrict__ in,
                                   __hip_bfloat16* __restrict__ out, int n) {
    int i = blockIdx.x * blockDim.x + threadIdx.x;
    if (i < n) out[i] = __float2bfloat16(in[i]);
}

// in [H, D, E] fp32  ->  out [H, E, D] bf16
__global__ void transpose_w_kernel(const float* __restrict__ in,
                                   __hip_bfloat16* __restrict__ out) {
    int i = blockIdx.x * blockDim.x + threadIdx.x;
    if (i >= HH * EE * DD) return;
    int d = i % DD;
    int t = i / DD;
    int e = t % EE;
    int h = t / EE;
    out[i] = __float2bfloat16(in[((size_t)h * DD + d) * EE + e]);
}

// ---------------- QKV projection (bf16 WMMA, fp32 acc) ----------------
__global__ __launch_bounds__(256) void qkv_proj_kernel(
    const __hip_bfloat16* __restrict__ xb,   // [B*S, D] bf16
    const __hip_bfloat16* __restrict__ Wt,   // [H, E, D] bf16 (transposed weight)
    const float* __restrict__ bias,          // [H, E] fp32
    __hip_bfloat16* __restrict__ out,
    float scale, int transposeOut)
{
    const int wave = blockIdx.x * 8 + (threadIdx.x >> 5);
    const int lane = threadIdx.x & 31;
    const int half = lane >> 4;
    const int l = lane & 15;
    const int bh   = wave >> 7;            // b*H + h   (0..31)
    const int srow = (wave & 127) << 4;    // row tile base
    const int b = bh >> 4, h = bh & 15;

    const __hip_bfloat16* Bm = Wt + (size_t)h * EE * DD;
    const int rowA = b * SS + srow;

    v8f zero = {0.f,0.f,0.f,0.f,0.f,0.f,0.f,0.f};
    v8f acc[4] = {zero, zero, zero, zero};

    for (int k0 = 0; k0 < DD; k0 += 32) {
        v16bf a = load_tile(xb, DD, rowA, k0);
#pragma unroll
        for (int j = 0; j < 4; ++j) {
            v16bf bm = load_tile(Bm, DD, j * 16, k0);
            acc[j] = wmma_bf16(a, bm, acc[j]);
        }
    }

#pragma unroll
    for (int j = 0; j < 4; ++j) {
        const int e = j * 16 + l;
        const float bv = bias[h * EE + e];
#pragma unroll
        for (int r = 0; r < 8; ++r) {
            const int m = srow + r + 8 * half;
            const float v = (acc[j][r] + bv) * scale;
            if (transposeOut)
                out[((size_t)bh * EE + e) * SS + m] = __float2bfloat16(v);
            else
                out[((size_t)bh * SS + m) * EE + e] = __float2bfloat16(v);
        }
    }
}

// ---------------- flash attention with TDM-staged K/V ----------------
// One block = 8 waves = 128 query rows of one (b,h). K/V kv-chunks (32 wide)
// are DMA'd into LDS by the Tensor Data Mover, double-buffered, shared by all
// 8 waves. Online softmax in fp32.
__global__ __launch_bounds__(256) void flash_attn_kernel(
    const __hip_bfloat16* __restrict__ Qb,   // [B*H, S, E]  (pre-scaled by 1/8)
    const __hip_bfloat16* __restrict__ Kb,   // [B*H, S, E]
    const __hip_bfloat16* __restrict__ Vt,   // [B*H, E, S]
    __hip_bfloat16* __restrict__ concat)     // [B, S, H*E]
{
    // K chunk: 32 rows x 64 bf16, padded to stride 66 (33 dwords) -> conflict-free
    // V chunk: 64 rows x 32 bf16, padded to stride 34 (17 dwords) -> conflict-free
    __shared__ __attribute__((aligned(16))) __hip_bfloat16 kbuf[2][32 * 66];
    __shared__ __attribute__((aligned(16))) __hip_bfloat16 vbuf[2][64 * 34];
    __shared__ __attribute__((aligned(16))) __hip_bfloat16 psh[8][16 * 34];

    const int wid  = threadIdx.x >> 5;
    const int lane = threadIdx.x & 31;
    const int half = lane >> 4;
    const int l = lane & 15;
    const int bh   = blockIdx.x >> 4;                       // one (b,h) per block
    const int srow = ((blockIdx.x & 15) * 8 + wid) << 4;    // this wave's 16 rows
    const int b = bh >> 4, h = bh & 15;
    __hip_bfloat16* myp = psh[wid];

    const __hip_bfloat16* Q  = Qb + (size_t)bh * SS * EE;
    const __hip_bfloat16* Kg = Kb + (size_t)bh * SS * EE;
    const __hip_bfloat16* Vg = Vt + (size_t)bh * EE * SS;

    // Q tile is loop-invariant: keep both K-halves resident.
    v16bf q0 = load_tile(Q, EE, srow, 0);
    v16bf q1 = load_tile(Q, EE, srow, 32);

    v8f zero = {0.f,0.f,0.f,0.f,0.f,0.f,0.f,0.f};
    v8f o[4] = {zero, zero, zero, zero};
    float rm[8], rs[8];
#pragma unroll
    for (int r = 0; r < 8; ++r) { rm[r] = -3.0e38f; rs[r] = 0.f; }

    // Prologue: DMA chunk 0 into buffer 0 (wave 0 only; TDM ignores EXEC).
    if (wid == 0) {
        tdm_load_2d_bf16(lds_offset(&kbuf[0][0]), Kg, EE, 32, EE, SS, EE, /*pad every 32 dw*/4, 0);
        tdm_load_2d_bf16(lds_offset(&vbuf[0][0]), Vg, 32, EE, SS, EE, SS, /*pad every 16 dw*/3, 0);
        __builtin_amdgcn_s_wait_tensorcnt(0);
    }
    __syncthreads();

    for (int c = 0; c < SS / 32; ++c) {
        const int cur = c & 1;
        const __hip_bfloat16* kc = kbuf[cur];
        const __hip_bfloat16* vc = vbuf[cur];

        // Kick off DMA of the next chunk into the other buffer (overlaps compute).
        if (wid == 0 && c + 1 < SS / 32) {
            const int t1 = (c + 1) * 32;
            tdm_load_2d_bf16(lds_offset(&kbuf[cur ^ 1][0]), Kg + (size_t)t1 * EE,
                             EE, 32, EE, SS, EE, 4, 0);
            tdm_load_2d_bf16(lds_offset(&vbuf[cur ^ 1][0]), Vg + t1,
                             32, EE, SS, EE, SS, 3, 0);
        }

        // scores: two 16x16 tiles covering kv [32c, 32c+32), operands from LDS
        v8f s0 = zero, s1 = zero;
        v16bf ka = load_tile(kc, 66, 0,  0);
        v16bf kb2 = load_tile(kc, 66, 16, 0);
        s0 = wmma_bf16(q0, ka,  s0);
        s1 = wmma_bf16(q0, kb2, s1);
        ka  = load_tile(kc, 66, 0,  32);
        kb2 = load_tile(kc, 66, 16, 32);
        s0 = wmma_bf16(q1, ka,  s0);
        s1 = wmma_bf16(q1, kb2, s1);

        // online softmax: rows r+8*half live across the 16 lanes of each half.
#pragma unroll
        for (int r = 0; r < 8; ++r) {
            float mx = fmaxf(s0[r], s1[r]);
            mx = fmaxf(mx, __shfl_xor(mx, 1, 32));
            mx = fmaxf(mx, __shfl_xor(mx, 2, 32));
            mx = fmaxf(mx, __shfl_xor(mx, 4, 32));
            mx = fmaxf(mx, __shfl_xor(mx, 8, 32));
            const float newm = fmaxf(rm[r], mx);
            const float corr = __expf(rm[r] - newm);
            rm[r] = newm;
            const float p0 = __expf(s0[r] - newm);
            const float p1 = __expf(s1[r] - newm);
            float ps = p0 + p1;
            ps += __shfl_xor(ps, 1, 32);
            ps += __shfl_xor(ps, 2, 32);
            ps += __shfl_xor(ps, 4, 32);
            ps += __shfl_xor(ps, 8, 32);
            rs[r] = rs[r] * corr + ps;
            o[0][r] *= corr; o[1][r] *= corr; o[2][r] *= corr; o[3][r] *= corr;
            const int row = r + 8 * half;
            myp[row * 34 + l]      = __float2bfloat16(p0);
            myp[row * 34 + 16 + l] = __float2bfloat16(p1);
        }

        // transpose P (C-layout -> A-layout) via per-wave LDS (same-wave dep)
        v16bf pA = load_tile(myp, 34, 0, 0);

        // O += P(16x32) @ V_chunk(32x64), V operand from LDS
#pragma unroll
        for (int j = 0; j < 4; ++j) {
            v16bf vb = load_tile(vc, 34, j * 16, 0);
            o[j] = wmma_bf16(pA, vb, o[j]);
        }

        // Publish next chunk: wave 0 drains TENSORcnt, then block-wide barrier.
        if (wid == 0) __builtin_amdgcn_s_wait_tensorcnt(0);
        __syncthreads();
    }

    // normalize and write directly into concat layout [B, S, H*E]
#pragma unroll
    for (int j = 0; j < 4; ++j) {
#pragma unroll
        for (int r = 0; r < 8; ++r) {
            const int m = srow + r + 8 * half;
            const float v = o[j][r] / rs[r];
            concat[((size_t)(b * SS + m) * HH + h) * EE + j * 16 + l] = __float2bfloat16(v);
        }
    }
}

// ---------------- output projection: concat @ Wp^T + bp (fp32 out) ----------------
__global__ __launch_bounds__(256) void out_proj_kernel(
    const __hip_bfloat16* __restrict__ Ab,   // [B*S, D] bf16
    const __hip_bfloat16* __restrict__ Wpb,  // [D, D] bf16, row n = output feature n
    const float* __restrict__ bp,            // [D]
    float* __restrict__ out)                 // [B*S, D] fp32
{
    const int wave = blockIdx.x * 8 + (threadIdx.x >> 5);
    const int lane = threadIdx.x & 31;
    const int half = lane >> 4;
    const int l = lane & 15;
    const int m0 = (wave >> 4) << 4;   // row tile
    const int n0 = (wave & 15) << 6;   // col group

    v8f zero = {0.f,0.f,0.f,0.f,0.f,0.f,0.f,0.f};
    v8f acc[4] = {zero, zero, zero, zero};

    for (int k0 = 0; k0 < DD; k0 += 32) {
        v16bf a = load_tile(Ab, DD, m0, k0);
#pragma unroll
        for (int j = 0; j < 4; ++j) {
            v16bf bm = load_tile(Wpb, DD, n0 + j * 16, k0);
            acc[j] = wmma_bf16(a, bm, acc[j]);
        }
    }

#pragma unroll
    for (int j = 0; j < 4; ++j) {
        const int n = n0 + j * 16 + l;
        const float bias = bp[n];
#pragma unroll
        for (int r = 0; r < 8; ++r) {
            out[(size_t)(m0 + r + 8 * half) * DD + n] = acc[j][r] + bias;
        }
    }
}

// ---------------- host launcher ----------------

extern "C" void kernel_launch(void* const* d_in, const int* in_sizes, int n_in,
                              void* d_out, int out_size, void* d_ws, size_t ws_size,
                              hipStream_t stream) {
    (void)in_sizes; (void)n_in; (void)out_size; (void)ws_size;
    const float* x  = (const float*)d_in[0];
    const float* Wq = (const float*)d_in[1];
    const float* Wk = (const float*)d_in[2];
    const float* Wv = (const float*)d_in[3];
    const float* bq = (const float*)d_in[4];
    const float* bk = (const float*)d_in[5];
    const float* bv = (const float*)d_in[6];
    const float* Wp = (const float*)d_in[7];
    const float* bp = (const float*)d_in[8];
    float* out = (float*)d_out;

    char* ws = (char*)d_ws;
    size_t off = 0;
    __hip_bfloat16* xb  = (__hip_bfloat16*)(ws + off); off += (size_t)BB * SS * DD * 2;       // 8 MB
    __hip_bfloat16* Wqt = (__hip_bfloat16*)(ws + off); off += (size_t)HH * EE * DD * 2;       // 2 MB
    __hip_bfloat16* Wkt = (__hip_bfloat16*)(ws + off); off += (size_t)HH * EE * DD * 2;
    __hip_bfloat16* Wvt = (__hip_bfloat16*)(ws + off); off += (size_t)HH * EE * DD * 2;
    __hip_bfloat16* Wpb = (__hip_bfloat16*)(ws + off); off += (size_t)DD * DD * 2;            // 2 MB
    __hip_bfloat16* Qb  = (__hip_bfloat16*)(ws + off); off += (size_t)BB * HH * SS * EE * 2;  // 8 MB
    __hip_bfloat16* Kb  = (__hip_bfloat16*)(ws + off); off += (size_t)BB * HH * SS * EE * 2;
    __hip_bfloat16* Vtb = (__hip_bfloat16*)(ws + off); off += (size_t)BB * HH * SS * EE * 2;
    __hip_bfloat16* Cb  = (__hip_bfloat16*)(ws + off); off += (size_t)BB * SS * DD * 2;       // 8 MB

    const int nx = BB * SS * DD;
    f32_to_bf16_kernel<<<(nx + 255) / 256, 256, 0, stream>>>(x, xb, nx);
    const int nw = HH * DD * EE;
    transpose_w_kernel<<<(nw + 255) / 256, 256, 0, stream>>>(Wq, Wqt);
    transpose_w_kernel<<<(nw + 255) / 256, 256, 0, stream>>>(Wk, Wkt);
    transpose_w_kernel<<<(nw + 255) / 256, 256, 0, stream>>>(Wv, Wvt);
    const int np = DD * DD;
    f32_to_bf16_kernel<<<(np + 255) / 256, 256, 0, stream>>>(Wp, Wpb, np);

    const int blocks = (BB * HH * (SS / 16)) / 8;  // 512 blocks x 8 waves
    qkv_proj_kernel<<<blocks, 256, 0, stream>>>(xb, Wqt, bq, Qb, 0.125f, 0); // fold 1/sqrt(64) into Q
    qkv_proj_kernel<<<blocks, 256, 0, stream>>>(xb, Wkt, bk, Kb, 1.0f, 0);
    qkv_proj_kernel<<<blocks, 256, 0, stream>>>(xb, Wvt, bv, Vtb, 1.0f, 1);  // V transposed [E,S]

    flash_attn_kernel<<<blocks, 256, 0, stream>>>(Qb, Kb, Vtb, Cb);

    const int blocks2 = ((BB * SS / 16) * (DD / 64)) / 8;  // 512
    out_proj_kernel<<<blocks2, 256, 0, stream>>>(Cb, Wpb, bp, out);
}
